// Capsule_65197603553898
// MI455X (gfx1250) — matrix operand order repaired and checked by reading, not compiled, not run on hardware
//
#include <hip/hip_runtime.h>
#include <hip/hip_bf16.h>

typedef _Float16 v16h __attribute__((ext_vector_type(16)));
typedef _Float16 v8h  __attribute__((ext_vector_type(8)));
typedef float    v8f  __attribute__((ext_vector_type(8)));

#define D_IN        1024
#define N_PER_B     512
#define NUM_CAP     10
#define DIM_CAP     16
#define N_OUT       (NUM_CAP * DIM_CAP)   // 160
#define M_TOTAL     (64 * N_PER_B)        // 32768
#define KCHUNK      32
#define NCHUNKS     (D_IN / KCHUNK)       // 32
#define WT_STRIDE   40                    // 32 k + 8 pad halves -> 80B col stride
#define WT_ELEMS    (N_OUT * WT_STRIDE)   // 6400 halves per buffer
#define UHAT_BYTES  ((size_t)64 * NUM_CAP * N_PER_B * DIM_CAP * sizeof(float))

// ---------------------------------------------------------------------------
// Kernel 0: one-time W transpose + f16 convert.
//   wt16[col][k] = (f16) W[k][col],  col in [0,160), k in [0,1024)
// 640 KB read once; makes GEMM chunk staging pure b128 copies.
// ---------------------------------------------------------------------------
__global__ void __launch_bounds__(256)
capsule_w_prep(const float* __restrict__ W, _Float16* __restrict__ wt16) {
    const int idx = blockIdx.x * 256 + threadIdx.x;   // < 160*1024
    const int col = idx >> 10;
    const int k   = idx & (D_IN - 1);
    wt16[col * D_IN + k] = (_Float16)W[k * N_OUT + col];
}

// ---------------------------------------------------------------------------
// Kernel 1: u_hat GEMM, single pass over x.
//   u_hat[b][cap][n][k] = sum_d x[b*512+n][d] * W[d][cap*16+k]
// Each wave owns one 16-row M tile and ALL 10 capsule output tiles: per
// 32-wide K chunk it builds one A fragment (f32 -> f16 in registers) and
// issues 10x v_wmma_f32_16x16x32_f16 against B fragments from an LDS-staged,
// double-buffered f16 W tile (vector b128 staging from the pre-transposed
// wt16, which stays L2-resident).
// ---------------------------------------------------------------------------
__global__ void __launch_bounds__(256)
capsule_uhat_gemm(const float* __restrict__ x, const _Float16* __restrict__ wt16,
                  float* __restrict__ uhat) {
    // W tile: [buf][col 0..159][kk 0..31], padded col stride. 2 x 12.5 KB.
    __shared__ _Float16 Wt[2][WT_ELEMS];

    const int tid  = threadIdx.x;
    const int wave = tid >> 5;
    const int lane = tid & 31;
    const int m0   = (blockIdx.x * 8 + wave) * 16;   // 16-row tile origin
    const int hs   = lane >> 4;                      // half-wave select
    const int mr   = lane & 15;                      // A row / B column (within tile)

    // A (16x32 f16): lanes 0-15 hold K = {0..7,16..23}; lanes 16-31 shifted +8.
    const float* __restrict__ arow = x + (size_t)(m0 + mr) * D_IN + hs * 8;
    // B (32x16 f16): lanes 0-15 hold K = {0..7,16..23}; lanes 16-31 shifted +8.
    const int boff = mr * WT_STRIDE + hs * 8;

    // Chunk staging: 160 cols x 32 halves = 640 b128 packets; thread t of
    // packet p: col = p>>2, quarter q = p&3 (8 halves each).
    // ---- stage chunk 0 into buffer 0 ----
#pragma unroll
    for (int j = 0; j < 3; ++j) {
        const int p = tid + j * 256;
        if (p < 640) {
            const int col = p >> 2, q = p & 3;
            *(v8h*)&Wt[0][col * WT_STRIDE + q * 8] =
                *(const v8h*)&wt16[col * D_IN + q * 8];
        }
    }
    __syncthreads();

    v8f acc[NUM_CAP];
#pragma unroll
    for (int i = 0; i < NUM_CAP; ++i) acc[i] = (v8f){};

#pragma unroll 1
    for (int c = 0; c < NCHUNKS; ++c) {
        const int cur = c & 1;

        // Prefetch next W chunk into registers (L2-resident).
        v8h wnext[3];
        if (c + 1 < NCHUNKS) {
#pragma unroll
            for (int j = 0; j < 3; ++j) {
                const int p = tid + j * 256;
                if (p < 640) {
                    const int col = p >> 2, q = p & 3;
                    wnext[j] = *(const v8h*)
                        &wt16[col * D_IN + (c + 1) * KCHUNK + q * 8];
                }
            }
        }

        // A fragment: f32 global loads -> f16 pack.
        const float* ap = arow + c * KCHUNK;
        if (c + 1 < NCHUNKS) __builtin_prefetch(ap + KCHUNK, 0, 0);
        const float4 a0 = *(const float4*)(ap + 0);
        const float4 a1 = *(const float4*)(ap + 4);
        const float4 a2 = *(const float4*)(ap + 16);
        const float4 a3 = *(const float4*)(ap + 20);
        v16h afrag;
        afrag[0]  = (_Float16)a0.x;  afrag[1]  = (_Float16)a0.y;
        afrag[2]  = (_Float16)a0.z;  afrag[3]  = (_Float16)a0.w;
        afrag[4]  = (_Float16)a1.x;  afrag[5]  = (_Float16)a1.y;
        afrag[6]  = (_Float16)a1.z;  afrag[7]  = (_Float16)a1.w;
        afrag[8]  = (_Float16)a2.x;  afrag[9]  = (_Float16)a2.y;
        afrag[10] = (_Float16)a2.z;  afrag[11] = (_Float16)a2.w;
        afrag[12] = (_Float16)a3.x;  afrag[13] = (_Float16)a3.y;
        afrag[14] = (_Float16)a3.z;  afrag[15] = (_Float16)a3.w;

        // 10 capsule tiles share this A fragment.
        const _Float16* bt = &Wt[cur][boff];
#pragma unroll
        for (int i = 0; i < NUM_CAP; ++i) {
            const _Float16* bp = bt + i * (DIM_CAP * WT_STRIDE);
            const v8h b0 = *(const v8h*)(bp + 0);
            const v8h b1 = *(const v8h*)(bp + 16);
            const v16h bfrag = __builtin_shufflevector(b0, b1,
                0, 1, 2, 3, 4, 5, 6, 7, 8, 9, 10, 11, 12, 13, 14, 15);
            acc[i] = __builtin_amdgcn_wmma_f32_16x16x32_f16(
                false, afrag, false, bfrag, (short)0, acc[i], false, false);
        }

        // Fill the other buffer for chunk c+1; single end-of-iteration barrier
        // orders these writes vs. next iteration's reads and next iteration's
        // writes vs. this iteration's reads.
        if (c + 1 < NCHUNKS) {
            _Float16* wdst = Wt[cur ^ 1];
#pragma unroll
            for (int j = 0; j < 3; ++j) {
                const int p = tid + j * 256;
                if (p < 640) {
                    const int col = p >> 2, q = p & 3;
                    *(v8h*)&wdst[col * WT_STRIDE + q * 8] = wnext[j];
                }
            }
        }
        __syncthreads();
    }

    // Store: D layout VGPR r -> (lanes 0-15: M=r, lanes 16-31: M=8+r), N = lane%16.
    const int b     = m0 >> 9;
    const int n0    = m0 & (N_PER_B - 1);
    const int rbase = hs * 8;
#pragma unroll
    for (int i = 0; i < NUM_CAP; ++i) {
        float* __restrict__ drow =
            uhat + (((size_t)b * NUM_CAP + i) * N_PER_B + n0) * DIM_CAP;
#pragma unroll
        for (int r = 0; r < 8; ++r)
            drow[(rbase + r) * DIM_CAP + mr] = acc[i][r];
    }
}

// ---------------------------------------------------------------------------
// Kernel 2: fused dynamic routing (3 iterations). One block per batch element.
// 320 threads = 10 waves, wave i owns capsule i. Logits/softmax live in LDS;
// u_hat re-reads hit L2 (20 MB << 192 MB).
// ---------------------------------------------------------------------------
__global__ void __launch_bounds__(320)
capsule_routing(const float* __restrict__ uhat, float* __restrict__ out) {
    __shared__ float b_lds[NUM_CAP * N_PER_B];   // 20 KB
    __shared__ float c_lds[NUM_CAP * N_PER_B];   // 20 KB
    __shared__ float out_lds[NUM_CAP * DIM_CAP];

    const int bidx = blockIdx.x;
    const int tid  = threadIdx.x;
    const int wave = tid >> 5;       // capsule index 0..9
    const int lane = tid & 31;
    const float* __restrict__ ucap =
        uhat + ((size_t)bidx * NUM_CAP + wave) * N_PER_B * DIM_CAP;

    for (int idx = tid; idx < NUM_CAP * N_PER_B; idx += 320) b_lds[idx] = 0.0f;
    __syncthreads();

    for (int iter = 0; iter < 3; ++iter) {
        // softmax over the capsule axis for each n
        for (int n = tid; n < N_PER_B; n += 320) {
            float m = b_lds[n];
#pragma unroll
            for (int i = 1; i < NUM_CAP; ++i) m = fmaxf(m, b_lds[i * N_PER_B + n]);
            float e[NUM_CAP];
            float s = 0.0f;
#pragma unroll
            for (int i = 0; i < NUM_CAP; ++i) {
                e[i] = __expf(b_lds[i * N_PER_B + n] - m);
                s += e[i];
            }
            const float inv = 1.0f / s;
#pragma unroll
            for (int i = 0; i < NUM_CAP; ++i) c_lds[i * N_PER_B + n] = e[i] * inv;
        }
        __syncthreads();

        // outputs[i][k] = sum_n c[i][n] * u_hat[i][n][k]
        float acc[DIM_CAP];
#pragma unroll
        for (int k = 0; k < DIM_CAP; ++k) acc[k] = 0.0f;
        for (int n = lane; n < N_PER_B; n += 32) {
            const float cv = c_lds[wave * N_PER_B + n];
            const float4* p = (const float4*)(ucap + n * DIM_CAP);
            const float4 q0 = p[0], q1 = p[1], q2 = p[2], q3 = p[3];
            acc[0]  += cv * q0.x; acc[1]  += cv * q0.y; acc[2]  += cv * q0.z; acc[3]  += cv * q0.w;
            acc[4]  += cv * q1.x; acc[5]  += cv * q1.y; acc[6]  += cv * q1.z; acc[7]  += cv * q1.w;
            acc[8]  += cv * q2.x; acc[9]  += cv * q2.y; acc[10] += cv * q2.z; acc[11] += cv * q2.w;
            acc[12] += cv * q3.x; acc[13] += cv * q3.y; acc[14] += cv * q3.z; acc[15] += cv * q3.w;
        }
#pragma unroll
        for (int off = 16; off >= 1; off >>= 1)
#pragma unroll
            for (int k = 0; k < DIM_CAP; ++k)
                acc[k] += __shfl_xor(acc[k], off, 32);

        if (lane == 0) {
            float ss = 1e-7f;
#pragma unroll
            for (int k = 0; k < DIM_CAP; ++k) ss += acc[k] * acc[k];
            const float inv = rsqrtf(ss);   // squash: x / sqrt(||x||^2 + eps)
#pragma unroll
            for (int k = 0; k < DIM_CAP; ++k) out_lds[wave * DIM_CAP + k] = acc[k] * inv;
        }
        __syncthreads();

        if (iter == 2) break;

        // b[i][n] = sum_k outputs[i][k] * u_hat[i][n][k]
        float ov[DIM_CAP];
#pragma unroll
        for (int k = 0; k < DIM_CAP; ++k) ov[k] = out_lds[wave * DIM_CAP + k];
        for (int n = lane; n < N_PER_B; n += 32) {
            const float4* p = (const float4*)(ucap + n * DIM_CAP);
            const float4 q0 = p[0], q1 = p[1], q2 = p[2], q3 = p[3];
            float d = ov[0]  * q0.x + ov[1]  * q0.y + ov[2]  * q0.z + ov[3]  * q0.w
                    + ov[4]  * q1.x + ov[5]  * q1.y + ov[6]  * q1.z + ov[7]  * q1.w
                    + ov[8]  * q2.x + ov[9]  * q2.y + ov[10] * q2.z + ov[11] * q2.w
                    + ov[12] * q3.x + ov[13] * q3.y + ov[14] * q3.z + ov[15] * q3.w;
            b_lds[wave * N_PER_B + n] = d;
        }
        __syncthreads();
    }

    if (tid < N_OUT) out[bidx * N_OUT + tid] = out_lds[tid];
}

extern "C" void kernel_launch(void* const* d_in, const int* in_sizes, int n_in,
                              void* d_out, int out_size, void* d_ws, size_t ws_size,
                              hipStream_t stream) {
    const float* x = (const float*)d_in[0];   // [64, 512, 1024] f32
    const float* W = (const float*)d_in[1];   // [1, 1024, 160]  f32
    float* out  = (float*)d_out;              // [64, 10, 16]    f32
    float* uhat = (float*)d_ws;               // [64, 10, 512, 16] f32 = 20 MB
    _Float16* wt16 = (_Float16*)((char*)d_ws + UHAT_BYTES);  // [160][1024] f16

    // W transpose + f16 convert: 160*1024 elements, 256/block.
    capsule_w_prep<<<(N_OUT * D_IN) / 256, 256, 0, stream>>>(W, wt16);

    // GEMM: 32768/16 = 2048 m-tiles, 8 waves/block -> 256 blocks; each wave
    // computes all 10 capsule tiles so x is streamed exactly once.
    capsule_uhat_gemm<<<M_TOTAL / (16 * 8), 256, 0, stream>>>(x, wt16, uhat);

    // Routing: one block per batch element, 10 waves (one per capsule).
    capsule_routing<<<64, 320, 0, stream>>>(uhat, out);
}
